// LLMToSDXLAdapter_78417512890389
// MI455X (gfx1250) — matrix operand order repaired and checked by reading, not compiled, not run on hardware
//
#include <hip/hip_runtime.h>
#include <math.h>

#define E 2048
#define NHEADS 16
#define HD 128
#define SLEN 512
#define BATCH 8
#define TGTQ 308
#define LLMD 1152
#define POOLD 1280
#define EPSLN 1e-5f

typedef __attribute__((ext_vector_type(16))) __bf16 v16bf;
typedef __attribute__((ext_vector_type(8)))  float  v8f;

union BF16Frag {
  v16bf v;
  unsigned int u[8];
};

// Pack two fp32 -> packed bf16 pair (round-to-nearest via +0x8000 bias, then
// byte-select the two high halves with a single v_perm_b32).
__device__ __forceinline__ unsigned int packbf(float a, float b) {
  unsigned int ua = __float_as_uint(a) + 0x8000u;
  unsigned int ub = __float_as_uint(b) + 0x8000u;
  return __builtin_amdgcn_perm(ub, ua, 0x07060302u);
}
__device__ __forceinline__ unsigned short f2bf(float a) {
  return (unsigned short)((__float_as_uint(a) + 0x8000u) >> 16);
}
__device__ __forceinline__ float geluf(float x) {
  return 0.5f * x * (1.0f + erff(x * 0.70710678118654752f));
}
__device__ __forceinline__ float rmax16(float v) {
  v = fmaxf(v, __shfl_xor(v, 1, 32));
  v = fmaxf(v, __shfl_xor(v, 2, 32));
  v = fmaxf(v, __shfl_xor(v, 4, 32));
  v = fmaxf(v, __shfl_xor(v, 8, 32));
  return v;
}
__device__ __forceinline__ float rsum16(float v) {
  v += __shfl_xor(v, 1, 32);
  v += __shfl_xor(v, 2, 32);
  v += __shfl_xor(v, 4, 32);
  v += __shfl_xor(v, 8, 32);
  return v;
}

// ---------------------------------------------------------------------------
// C[M,N] = epi(A[M,K] @ W[N,K]^T + bias [+ pos]) [+ residual]
// epi: 0 none, 1 gelu, 2 sigmoid.  Tile: 128(M) x 64(N) x 32(K), 8 waves,
// double-buffered LDS so tile k+1 global loads overlap tile k WMMAs.
// ---------------------------------------------------------------------------
__global__ __launch_bounds__(256)
void gemm_wmma(const float* __restrict__ A, const float* __restrict__ W,
               const float* __restrict__ bias, const float* __restrict__ residual,
               const float* __restrict__ pos, int posPeriod,
               float* __restrict__ C, int M, int N, int K, int epi) {
  __shared__ __attribute__((aligned(16))) unsigned short As[2][128 * 40];
  __shared__ __attribute__((aligned(16))) unsigned short Bs[2][64 * 40];

  const int tid  = threadIdx.x;
  const int lane = tid & 31;
  const int wid  = tid >> 5;
  const int wm   = wid >> 1;        // 0..3 -> 32-row slab
  const int wn   = wid & 1;         // 0..1 -> 32-col slab
  const int half = lane >> 4;
  const int l15  = lane & 15;

  const int tm0 = blockIdx.y * 128;
  const int tn0 = blockIdx.x * 64;

  const v8f vzero = {0.f,0.f,0.f,0.f,0.f,0.f,0.f,0.f};
  v8f acc[2][2];
  acc[0][0] = vzero; acc[0][1] = vzero; acc[1][0] = vzero; acc[1][1] = vzero;

  const int arow = tid >> 1, aseg = tid & 1;   // A: 128 rows x 32k, 16 floats/thread
  const int brow = tid >> 2, bseg = tid & 3;   // W:  64 rows x 32k,  8 floats/thread
  const int ntiles = K >> 5;

  auto stageA = [&](int kt, int buf) {
    const int gm = tm0 + arow;
    unsigned int u[8];
    if (gm < M) {
      const float* p = A + (size_t)gm * K + kt * 32 + aseg * 16;
      float4 x0 = ((const float4*)p)[0];
      float4 x1 = ((const float4*)p)[1];
      float4 x2 = ((const float4*)p)[2];
      float4 x3 = ((const float4*)p)[3];
      u[0]=packbf(x0.x,x0.y); u[1]=packbf(x0.z,x0.w);
      u[2]=packbf(x1.x,x1.y); u[3]=packbf(x1.z,x1.w);
      u[4]=packbf(x2.x,x2.y); u[5]=packbf(x2.z,x2.w);
      u[6]=packbf(x3.x,x3.y); u[7]=packbf(x3.z,x3.w);
      if (kt + 1 < ntiles) __builtin_prefetch(p + 32, 0, 1);
    } else {
      #pragma unroll
      for (int i = 0; i < 8; i++) u[i] = 0u;
    }
    unsigned int* d = (unsigned int*)(As[buf] + arow * 40 + aseg * 16);
    #pragma unroll
    for (int i = 0; i < 8; i++) d[i] = u[i];
  };
  auto stageB = [&](int kt, int buf) {
    const int gn = tn0 + brow;
    unsigned int u[4];
    if (gn < N) {
      const float* p = W + (size_t)gn * K + kt * 32 + bseg * 8;
      float4 x0 = ((const float4*)p)[0];
      float4 x1 = ((const float4*)p)[1];
      u[0]=packbf(x0.x,x0.y); u[1]=packbf(x0.z,x0.w);
      u[2]=packbf(x1.x,x1.y); u[3]=packbf(x1.z,x1.w);
      if (kt + 1 < ntiles) __builtin_prefetch(p + 32, 0, 1);
    } else {
      u[0]=u[1]=u[2]=u[3]=0u;
    }
    unsigned int* d = (unsigned int*)(Bs[buf] + brow * 40 + bseg * 8);
    d[0]=u[0]; d[1]=u[1]; d[2]=u[2]; d[3]=u[3];
  };

  stageA(0, 0);
  stageB(0, 0);
  int p = 0;
  for (int kt = 0; kt < ntiles; kt++) {
    __syncthreads();                    // buf p ready; prior reads of buf p^1 done
    if (kt + 1 < ntiles) {              // overlap next tile's loads with WMMAs
      stageA(kt + 1, p ^ 1);
      stageB(kt + 1, p ^ 1);
    }
    BF16Frag fa[2], fb[2];
    #pragma unroll
    for (int mt = 0; mt < 2; mt++) {    // A frag: k = j + 16*half
      const unsigned int* q = (const unsigned int*)(As[p] + (wm*32 + mt*16 + l15) * 40);
      #pragma unroll
      for (int i = 0; i < 8; i++) fa[mt].u[i] = q[8*half + i];
    }
    #pragma unroll
    for (int nt = 0; nt < 2; nt++) {    // B frag: k = j + 16*half
      const unsigned int* q = (const unsigned int*)(Bs[p] + (wn*32 + nt*16 + l15) * 40);
      #pragma unroll
      for (int i = 0; i < 8; i++) fb[nt].u[i] = q[8*half + i];
    }
    #pragma unroll
    for (int mt = 0; mt < 2; mt++)
      #pragma unroll
      for (int nt = 0; nt < 2; nt++)
        acc[mt][nt] = __builtin_amdgcn_wmma_f32_16x16x32_bf16(
            false, fa[mt].v, false, fb[nt].v, (short)0, acc[mt][nt], false, false);
    p ^= 1;
  }

  #pragma unroll
  for (int mt = 0; mt < 2; mt++) {
    #pragma unroll
    for (int nt = 0; nt < 2; nt++) {
      const int n = tn0 + wn*32 + nt*16 + l15;
      #pragma unroll
      for (int r = 0; r < 8; r++) {
        const int m = tm0 + wm*32 + mt*16 + half*8 + r;
        if (m < M && n < N) {
          float v = acc[mt][nt][r] + bias[n];
          if (pos) v += pos[(size_t)(m % posPeriod) * N + n];
          if (epi == 1) v = geluf(v);
          else if (epi == 2) v = 1.0f / (1.0f + __expf(-v));
          if (residual) v += residual[(size_t)m * N + n];
          C[(size_t)m * N + n] = v;
        }
      }
    }
  }
}

// ---------------------------------------------------------------------------
// LayerNorm over E=2048, one block (256 thr) per row. Optional pos-add / GELU.
// In-place safe (row kept in registers).
// ---------------------------------------------------------------------------
__global__ __launch_bounds__(256)
void layernorm_k(const float* __restrict__ X, const float* __restrict__ g,
                 const float* __restrict__ b, const float* __restrict__ pos,
                 int posPeriod, int doGelu, float* __restrict__ Y, int M) {
  const int row = blockIdx.x;
  const int tid = threadIdx.x;
  if (row >= M) return;
  const float* xr = X + (size_t)row * E;
  float v[8], s1 = 0.f, s2 = 0.f;
  #pragma unroll
  for (int i = 0; i < 8; i++) {
    v[i] = xr[tid + i * 256];
    s1 += v[i]; s2 += v[i] * v[i];
  }
  __shared__ float r1[256], r2[256];
  r1[tid] = s1; r2[tid] = s2;
  __syncthreads();
  for (int off = 128; off > 0; off >>= 1) {
    if (tid < off) { r1[tid] += r1[tid+off]; r2[tid] += r2[tid+off]; }
    __syncthreads();
  }
  const float mean = r1[0] * (1.0f / E);
  const float var  = r2[0] * (1.0f / E) - mean * mean;
  const float rs   = rsqrtf(var + EPSLN);
  float* yr = Y + (size_t)row * E;
  const float* pr = pos ? pos + (size_t)(row % posPeriod) * E : nullptr;
  #pragma unroll
  for (int i = 0; i < 8; i++) {
    const int c = tid + i * 256;
    float y = (v[i] - mean) * rs * g[c] + b[c];
    if (pr) y += pr[c];
    if (doGelu) y = geluf(y);
    yr[c] = y;
  }
}

// ---------------------------------------------------------------------------
// Flash attention, one wave per (b, h, 16-query tile). QKV packed [*,3E].
// Q col off 0, K at +E, V at +2E. qBS==0 => broadcast queries across batch.
// ---------------------------------------------------------------------------
__global__ __launch_bounds__(32)
void attn_wmma(const float* __restrict__ Qb, long long qBS,
               const float* __restrict__ KVb, long long kvBS,
               const int* __restrict__ mask,
               float* __restrict__ O, int Lq, int Lk, float scale) {
  const int lane = threadIdx.x;
  const int half = lane >> 4;
  const int l15  = lane & 15;
  const int qt = blockIdx.x, hh = blockIdx.y, bb = blockIdx.z;

  __shared__ __attribute__((aligned(16))) unsigned short Ps[16 * 40];

  int qrow = qt * 16 + l15;
  if (qrow >= Lq) qrow = Lq - 1;
  const float* qbase = Qb + (size_t)bb * qBS + (size_t)qrow * (3 * E) + hh * HD;

  BF16Frag qf[4];                      // Q as A operand: 4 chunks of k=32 over HD
  #pragma unroll
  for (int c = 0; c < 4; c++) {
    const float* p = qbase + c * 32;
    float4 x0 = ((const float4*)(p + 16 * half))[0];
    float4 x1 = ((const float4*)(p + 16 * half))[1];
    float4 x2 = ((const float4*)(p + 16 * half))[2];
    float4 x3 = ((const float4*)(p + 16 * half))[3];
    qf[c].u[0]=packbf(x0.x,x0.y); qf[c].u[1]=packbf(x0.z,x0.w);
    qf[c].u[2]=packbf(x1.x,x1.y); qf[c].u[3]=packbf(x1.z,x1.w);
    qf[c].u[4]=packbf(x2.x,x2.y); qf[c].u[5]=packbf(x2.z,x2.w);
    qf[c].u[6]=packbf(x3.x,x3.y); qf[c].u[7]=packbf(x3.z,x3.w);
  }

  const v8f vzero = {0.f,0.f,0.f,0.f,0.f,0.f,0.f,0.f};
  v8f o[8];
  #pragma unroll
  for (int i = 0; i < 8; i++) o[i] = vzero;
  float ms[8], ls[8];
  #pragma unroll
  for (int r = 0; r < 8; r++) { ms[r] = -1e30f; ls[r] = 0.f; }

  const float* Kbase = KVb + (size_t)bb * kvBS + E + hh * HD;
  const float* Vbase = KVb + (size_t)bb * kvBS + 2 * E + hh * HD;

  for (int k0 = 0; k0 < Lk; k0 += 32) {
    v8f s0 = vzero, s1 = vzero;
    #pragma unroll
    for (int t = 0; t < 2; t++) {
      int key = k0 + t * 16 + l15;
      int keyc = key < Lk ? key : Lk - 1;
      const float* kb = Kbase + (size_t)keyc * (3 * E);
      #pragma unroll
      for (int c = 0; c < 4; c++) {    // K^T as B operand: k = j + 16*half
        BF16Frag kf;
        const float* p = kb + c * 32 + 16 * half;
        float4 y0 = ((const float4*)p)[0];
        float4 y1 = ((const float4*)p)[1];
        float4 y2 = ((const float4*)p)[2];
        float4 y3 = ((const float4*)p)[3];
        kf.u[0]=packbf(y0.x,y0.y); kf.u[1]=packbf(y0.z,y0.w);
        kf.u[2]=packbf(y1.x,y1.y); kf.u[3]=packbf(y1.z,y1.w);
        kf.u[4]=packbf(y2.x,y2.y); kf.u[5]=packbf(y2.z,y2.w);
        kf.u[6]=packbf(y3.x,y3.y); kf.u[7]=packbf(y3.z,y3.w);
        if (t == 0)
          s0 = __builtin_amdgcn_wmma_f32_16x16x32_bf16(false, qf[c].v, false, kf.v, (short)0, s0, false, false);
        else
          s1 = __builtin_amdgcn_wmma_f32_16x16x32_bf16(false, qf[c].v, false, kf.v, (short)0, s1, false, false);
      }
    }
    const int key0 = k0 + l15, key1 = k0 + 16 + l15;
    const bool v0 = (key0 < Lk) && (!mask || mask[(size_t)bb * Lk + key0] != 0);
    const bool v1 = (key1 < Lk) && (!mask || mask[(size_t)bb * Lk + key1] != 0);
    float pr0[8], pr1[8], alpha[8];
    #pragma unroll
    for (int r = 0; r < 8; r++) {
      float a = v0 ? s0[r] * scale : -1e30f;
      float bsc = v1 ? s1[r] * scale : -1e30f;
      float rowm = rmax16(fmaxf(a, bsc));
      float nm = fmaxf(ms[r], rowm);
      pr0[r] = __expf(a - nm);
      pr1[r] = __expf(bsc - nm);
      float rowsum = rsum16(pr0[r] + pr1[r]);
      alpha[r] = __expf(ms[r] - nm);
      ls[r] = ls[r] * alpha[r] + rowsum;
      ms[r] = nm;
    }
    #pragma unroll
    for (int nt = 0; nt < 8; nt++)
      #pragma unroll
      for (int r = 0; r < 8; r++)
        o[nt][r] *= alpha[r];

    // D-layout probs -> LDS -> A-layout fragment
    #pragma unroll
    for (int r = 0; r < 8; r++) {
      Ps[(half * 8 + r) * 40 + l15]      = f2bf(pr0[r]);
      Ps[(half * 8 + r) * 40 + 16 + l15] = f2bf(pr1[r]);
    }
    __syncthreads();
    BF16Frag pf;
    {
      const unsigned int* pp = (const unsigned int*)(Ps + l15 * 40);
      #pragma unroll
      for (int i = 0; i < 8; i++) pf.u[i] = pp[8*half + i];
    }
    __syncthreads();

    #pragma unroll
    for (int nt = 0; nt < 8; nt++) {   // V as B operand: element j -> key k0+j+16*half
      const float* vb = Vbase + nt * 16 + l15;
      BF16Frag vf;
      #pragma unroll
      for (int i = 0; i < 8; i++) {
        int ka = k0 + 16 * half + 2 * i;
        int kb2 = ka + 1;
        if (ka  >= Lk) ka  = Lk - 1;
        if (kb2 >= Lk) kb2 = Lk - 1;
        vf.u[i] = packbf(vb[(size_t)ka * (3 * E)], vb[(size_t)kb2 * (3 * E)]);
      }
      o[nt] = __builtin_amdgcn_wmma_f32_16x16x32_bf16(false, pf.v, false, vf.v, (short)0, o[nt], false, false);
    }
  }

  #pragma unroll
  for (int nt = 0; nt < 8; nt++) {
    #pragma unroll
    for (int r = 0; r < 8; r++) {
      const int q = qt * 16 + half * 8 + r;
      if (q < Lq)
        O[((size_t)bb * Lq + q) * E + hh * HD + nt * 16 + l15] = o[nt][r] / ls[r];
    }
  }
}

// ---------------------------------------------------------------------------
__global__ void concat_gate_in(const float* __restrict__ cq, const float* __restrict__ comp0,
                               float* __restrict__ out, int M) {
  int idx = blockIdx.x * blockDim.x + threadIdx.x;
  if (idx >= M * 2 * E) return;
  int m = idx / (2 * E), c = idx % (2 * E);
  out[idx] = (c < E) ? cq[(size_t)(m % TGTQ) * E + c] : comp0[(size_t)m * E + (c - E)];
}

__global__ void gate_combine(const float* __restrict__ g, const float* __restrict__ comp0,
                             const float* __restrict__ cq, float* __restrict__ out, int M) {
  int idx = blockIdx.x * blockDim.x + threadIdx.x;
  if (idx >= M * E) return;
  int m = idx / E, c = idx % E;
  float gg = g[idx];
  out[idx] = gg * comp0[idx] + (1.f - gg) * cq[(size_t)(m % TGTQ) * E + c];
}

// ---------------------------------------------------------------------------
extern "C" void kernel_launch(void* const* d_in, const int* in_sizes, int n_in,
                              void* d_out, int out_size, void* d_ws, size_t ws_size,
                              hipStream_t stream) {
  (void)in_sizes; (void)n_in; (void)out_size; (void)ws_size;
  const float* llm    = (const float*)d_in[0];
  const int*   amask  = (const int*)d_in[1];
  const float* proj_w = (const float*)d_in[2];
  const float* proj_b = (const float*)d_in[3];
  const float* in_pos = (const float*)d_in[4];
  const float* out_pos= (const float*)d_in[5];
  const float *w_ln1g=(const float*)d_in[6],  *w_ln1b=(const float*)d_in[7];
  const float *w_aiw =(const float*)d_in[8],  *w_aib =(const float*)d_in[9];
  const float *w_aow =(const float*)d_in[10], *w_aob =(const float*)d_in[11];
  const float *w_ln2g=(const float*)d_in[12], *w_ln2b=(const float*)d_in[13];
  const float *w_m1w =(const float*)d_in[14], *w_m1b =(const float*)d_in[15];
  const float *w_m2w =(const float*)d_in[16], *w_m2b =(const float*)d_in[17];
  const float *n_ln1g=(const float*)d_in[18], *n_ln1b=(const float*)d_in[19];
  const float *n_aiw =(const float*)d_in[20], *n_aib =(const float*)d_in[21];
  const float *n_aow =(const float*)d_in[22], *n_aob =(const float*)d_in[23];
  const float *n_ln2g=(const float*)d_in[24], *n_ln2b=(const float*)d_in[25];
  const float *n_m1w =(const float*)d_in[26], *n_m1b =(const float*)d_in[27];
  const float *n_m2w =(const float*)d_in[28], *n_m2b =(const float*)d_in[29];
  const float* cq    = (const float*)d_in[30];
  const float* c_aiw = (const float*)d_in[31];
  const float* c_aib = (const float*)d_in[32];
  const float* c_aow = (const float*)d_in[33];
  const float* c_aob = (const float*)d_in[34];
  const float* c_ng  = (const float*)d_in[35];
  const float* c_nb  = (const float*)d_in[36];
  const float* gate_w= (const float*)d_in[37];
  const float* gate_b= (const float*)d_in[38];
  const float* pool_tok=(const float*)d_in[39];
  const float* p_aiw = (const float*)d_in[40];
  const float* p_aib = (const float*)d_in[41];
  const float* p_aow = (const float*)d_in[42];
  const float* p_aob = (const float*)d_in[43];
  const float* pp_w1 = (const float*)d_in[44];
  const float* pp_b1 = (const float*)d_in[45];
  const float* pp_lng= (const float*)d_in[46];
  const float* pp_lnb= (const float*)d_in[47];
  const float* pp_w2 = (const float*)d_in[48];
  const float* pp_b2 = (const float*)d_in[49];

  const int M1 = BATCH * SLEN;   // 4096
  const int M2 = BATCH * TGTQ;   // 2464
  const float SDPA = 0.088388347648318447f;   // 1/sqrt(128)
  const float CROSS = 1.0f / 0.7f;

  float* ws  = (float*)d_ws;
  float* x   = ws;
  float* h   = x  + (size_t)M1 * E;
  float* t0  = h  + (size_t)M1 * E;
  float* big = t0 + (size_t)M1 * E;                 // M1 x 4E (also QKV, gate-cat)
  float* qq  = big + (size_t)M1 * 4 * E;            // TGT x 3E
  float* pq  = qq  + (size_t)TGTQ * 3 * E;          // 1 x 3E

  auto gemm = [&](const float* A, const float* W, const float* bias,
                  const float* res, const float* pos, int posP,
                  float* C, int M, int N, int K, int epi) {
    dim3 g((N + 63) / 64, (M + 127) / 128);
    gemm_wmma<<<g, 256, 0, stream>>>(A, W, bias, res, pos, posP, C, M, N, K, epi);
  };
  auto ln = [&](const float* X, const float* gam, const float* bet,
                const float* pos, int posP, int gelu, float* Y, int M) {
    layernorm_k<<<M, 256, 0, stream>>>(X, gam, bet, pos, posP, gelu, Y, M);
  };
  auto attn = [&](const float* Qb, long long qBS, const float* KVb, long long kvBS,
                  const int* mk, float* Op, int Lq, int Lk, float scale) {
    dim3 g((Lq + 15) / 16, NHEADS, BATCH);
    attn_wmma<<<g, 32, 0, stream>>>(Qb, qBS, KVb, kvBS, mk, Op, Lq, Lk, scale);
  };

  // ---- input projection: x = llm @ proj_w.T + proj_b + in_pos ----
  gemm(llm, proj_w, proj_b, nullptr, in_pos, SLEN, x, M1, E, LLMD, 0);

  // ---- encoder blocks ----
  const long long bs3e = (long long)SLEN * 3 * E;
  for (int i = 0; i < 3; i++) {
    ln(x, w_ln1g + i*E, w_ln1b + i*E, nullptr, 1, 0, h, M1);
    gemm(h, w_aiw + (size_t)i*3*E*E, w_aib + (size_t)i*3*E, nullptr, nullptr, 1, big, M1, 3*E, E, 0);
    attn(big, bs3e, big, bs3e, amask, t0, SLEN, SLEN, SDPA);
    gemm(t0, w_aow + (size_t)i*E*E, w_aob + (size_t)i*E, x, nullptr, 1, x, M1, E, E, 0);
    ln(x, w_ln2g + i*E, w_ln2b + i*E, nullptr, 1, 0, h, M1);
    gemm(h, w_m1w + (size_t)i*4*E*E, w_m1b + (size_t)i*4*E, nullptr, nullptr, 1, big, M1, 4*E, E, 1);
    gemm(big, w_m2w + (size_t)i*E*4*E, w_m2b + (size_t)i*E, x, nullptr, 1, x, M1, E, 4*E, 0);
  }

  // ---- sharpened cross-attention ----
  gemm(cq, c_aiw, c_aib, nullptr, nullptr, 1, qq, TGTQ, 3*E, E, 0);     // queries (broadcast)
  gemm(x,  c_aiw, c_aib, nullptr, nullptr, 1, big, M1, 3*E, E, 0);      // K/V from x
  attn(qq, 0LL, big, bs3e, amask, t0, TGTQ, SLEN, CROSS);
  gemm(t0, c_aow, c_aob, nullptr, nullptr, 1, h, M2, E, E, 0);          // comp0 -> h
  {
    int tot = M2 * 2 * E;
    concat_gate_in<<<(tot + 255) / 256, 256, 0, stream>>>(cq, h, big, M2);
  }
  gemm(big, gate_w, gate_b, nullptr, nullptr, 1, t0, M2, E, 2*E, 2);    // sigmoid gate -> t0
  {
    int tot = M2 * E;
    gate_combine<<<(tot + 255) / 256, 256, 0, stream>>>(t0, h, cq, x, M2);
  }
  ln(x, c_ng, c_nb, out_pos, TGTQ, 0, x, M2);                           // LN + out_pos

  // ---- neck blocks ----
  const long long bs3t = (long long)TGTQ * 3 * E;
  for (int i = 0; i < 3; i++) {
    ln(x, n_ln1g + i*E, n_ln1b + i*E, nullptr, 1, 0, h, M2);
    gemm(h, n_aiw + (size_t)i*3*E*E, n_aib + (size_t)i*3*E, nullptr, nullptr, 1, big, M2, 3*E, E, 0);
    attn(big, bs3t, big, bs3t, nullptr, t0, TGTQ, TGTQ, SDPA);
    gemm(t0, n_aow + (size_t)i*E*E, n_aob + (size_t)i*E, x, nullptr, 1, x, M2, E, E, 0);
    ln(x, n_ln2g + i*E, n_ln2b + i*E, nullptr, 1, 0, h, M2);
    gemm(h, n_m1w + (size_t)i*4*E*E, n_m1b + (size_t)i*4*E, nullptr, nullptr, 1, big, M2, 4*E, E, 1);
    gemm(big, n_m2w + (size_t)i*E*4*E, n_m2b + (size_t)i*E, x, nullptr, 1, x, M2, E, 4*E, 0);
  }

  // ---- attention pooling + head ----
  gemm(pool_tok, p_aiw, p_aib, nullptr, nullptr, 1, pq, 1, 3*E, E, 0);
  gemm(x, p_aiw, p_aib, nullptr, nullptr, 1, big, M2, 3*E, E, 0);
  attn(pq, 0LL, big, bs3t, nullptr, t0, 1, TGTQ, SDPA);                 // -> [B, E]
  gemm(t0, p_aow, p_aob, nullptr, nullptr, 1, h, BATCH, E, E, 0);       // pooled
  gemm(h, pp_w1, pp_b1, nullptr, nullptr, 1, t0, BATCH, E, E, 0);
  ln(t0, pp_lng, pp_lnb, nullptr, 1, 1, t0, BATCH);                     // LN then GELU
  gemm(t0, pp_w2, pp_b2, nullptr, nullptr, 1, (float*)d_out + (size_t)M2 * E,
       BATCH, POOLD, E, 0);

  // ---- outputs: [comp (B*TGT*E), head (B*POOLD)] ----
  hipMemcpyAsync(d_out, x, (size_t)M2 * E * sizeof(float),
                 hipMemcpyDeviceToDevice, stream);
}
